// Attention_58548994179786
// MI455X (gfx1250) — compile-verified, matrix operand order checked by reading, not run on hardware
//
#include <hip/hip_runtime.h>
#include <hip/hip_bf16.h>

typedef __attribute__((ext_vector_type(16))) _Float16 v16h;
typedef __attribute__((ext_vector_type(8)))  _Float16 h8;
typedef __attribute__((ext_vector_type(8)))  float    v8f;
typedef __attribute__((ext_vector_type(4)))  float    f4;
typedef __attribute__((ext_vector_type(4)))  int      v4i;

#define N_HEAD 8
#define D_HID 32
#define D_MODEL 256
#define SEQ 2048
#define BATCH 2
#define NROWS (BATCH * SEQ)   // 4096
#define LDT 48                // LDS tile pitch in halves (96B: 16B aligned, conflict-reduced)

// ---------- CDNA5 async Global->LDS path (probe via __has_builtin; safe fallback) ----------
#if __has_builtin(__builtin_amdgcn_global_load_async_to_lds_b128)
#define ASYNC_LDS 1
#if __has_builtin(__builtin_amdgcn_s_wait_asynccnt)
#define WAIT_ASYNC(n) __builtin_amdgcn_s_wait_asynccnt(n)
#else
#define WAIT_ASYNC(n) asm volatile("s_wait_asynccnt %0" :: "i"(n) : "memory")
#endif
#else
#define WAIT_ASYNC(n) ((void)0)
#endif

__device__ __forceinline__ void cp16_to_lds(const _Float16* g, _Float16* l) {
#ifdef ASYNC_LDS
    // signature (from hipcc diagnostic): (v4i addrspace(1)*, v4i addrspace(3)*, imm, imm)
    __builtin_amdgcn_global_load_async_to_lds_b128(
        (__attribute__((address_space(1))) v4i*)g,
        (__attribute__((address_space(3))) v4i*)l, 0, 0);
#else
    *(h8*)l = *(const h8*)g;   // sync fallback: global_load_b128 + ds_store_b128
#endif
}

// ---------- WMMA wrapper ----------
__device__ __forceinline__ v8f wmma_f16(v16h a, v16h b, v8f c) {
    // D = A(16x32 f16) * B(32x16 f16) + C(16x16 f32)
    return __builtin_amdgcn_wmma_f32_16x16x32_f16(false, a, false, b, (short)0, c, false, false);
}

// ---------- Fragment loaders ----------
// A-frag (16x32) from row-major f16 [ld]: lane holds row lane&15.
// Same code loads a B-frag (32x16) from a row-major [N,K] matrix (lane holds col n).
// K(half j, hi) = (j&7) + 8*hi + 16*(j>>3)  ==> two contiguous 8-half (16B) loads.
__device__ __forceinline__ v16h load_frag_h16(const _Float16* __restrict__ base, int ld, int lane) {
    int lr = lane & 15, hw = lane >> 4;
    const _Float16* p = base + (long)lr * ld + 8 * hw;
    h8 lo = *(const h8*)(p);
    h8 hh = *(const h8*)(p + 16);
    v16h r;
#pragma unroll
    for (int i = 0; i < 8; ++i) { r[i] = lo[i]; r[i + 8] = hh[i]; }
    return r;
}

// Same fragment but sourced from a row-major fp32 matrix, converting (and scaling) to f16.
__device__ __forceinline__ v16h load_frag_f32(const float* __restrict__ base, int ld, int lane, float scale) {
    int lr = lane & 15, hw = lane >> 4;
    const float* p = base + (long)lr * ld + 8 * hw;
    f4 a0 = *(const f4*)(p);
    f4 a1 = *(const f4*)(p + 4);
    f4 a2 = *(const f4*)(p + 16);
    f4 a3 = *(const f4*)(p + 20);
    v16h r;
#pragma unroll
    for (int i = 0; i < 4; ++i) {
        r[i]      = (_Float16)(a0[i] * scale);
        r[4 + i]  = (_Float16)(a1[i] * scale);
        r[8 + i]  = (_Float16)(a2[i] * scale);
        r[12 + i] = (_Float16)(a3[i] * scale);
    }
    return r;
}

// ---------- Cross-lane row reductions over the 16 lanes holding one C row ----------
__device__ __forceinline__ float rowmax16(float v) {
#pragma unroll
    for (int off = 1; off < 16; off <<= 1) v = fmaxf(v, __shfl_xor(v, off, 32));
    return v;
}
__device__ __forceinline__ float rowsum16(float v) {
#pragma unroll
    for (int off = 1; off < 16; off <<= 1) v += __shfl_xor(v, off, 32);
    return v;
}

// ---------- Kernel 0: weight fp32 -> f16 ----------
__global__ __launch_bounds__(256) void attn_cvt_weights(
    const float* __restrict__ wq, const float* __restrict__ wk,
    const float* __restrict__ wv, const float* __restrict__ wo,
    _Float16* __restrict__ out) {
    int i = blockIdx.x * 256 + threadIdx.x;           // 65536 threads
    out[i]          = (_Float16)wq[i];
    out[65536 + i]  = (_Float16)wk[i];
    out[131072 + i] = (_Float16)wv[i];
    out[196608 + i] = (_Float16)wo[i];
}

// ---------- Kernel 1: projection  out = X @ W^T  (f16 result, head-major) ----------
// transposed==0 : out[((b*8+h)*2048 + s)*32 + d]         (Q, K)
// transposed==1 : out[((b*8+h)*32 + d)*2048 + s]         (V, pre-transposed for P*V)
__global__ __launch_bounds__(256) void attn_proj(
    const float* __restrict__ X, const _Float16* __restrict__ W16,
    _Float16* __restrict__ out, int transposed, float scale) {
    int lane = threadIdx.x & 31;
    int wave = threadIdx.x >> 5;        // 8 waves
    int row0 = blockIdx.x * 16;         // 256 blocks cover 4096 rows
    int n0 = wave * 32;                 // each wave: 32 output cols (2 WMMA N-tiles)
    v8f c0 = {}, c1 = {};
#pragma unroll
    for (int k0 = 0; k0 < D_MODEL; k0 += 32) {
        v16h af = load_frag_f32(X + (long)row0 * D_MODEL + k0, D_MODEL, lane, scale);
        v16h b0 = load_frag_h16(W16 + (long)n0 * D_MODEL + k0, D_MODEL, lane);
        v16h b1 = load_frag_h16(W16 + (long)(n0 + 16) * D_MODEL + k0, D_MODEL, lane);
        c0 = wmma_f16(af, b0, c0);
        c1 = wmma_f16(af, b1, c1);
    }
    int lr = lane & 15, hw = lane >> 4;
#pragma unroll
    for (int r = 0; r < 8; ++r) {
        int m = row0 + r + 8 * hw;      // global row
        int b = m >> 11, s = m & 2047;
#pragma unroll
        for (int t = 0; t < 2; ++t) {
            int col = n0 + 16 * t + lr;
            int h = col >> 5, d = col & 31;
            float v = t ? c1[r] : c0[r];
            long idx = transposed ? (((long)(b * N_HEAD + h) * D_HID + d) * SEQ + s)
                                  : (((long)(b * N_HEAD + h) * SEQ + s) * D_HID + d);
            out[idx] = (_Float16)v;
        }
    }
}

// ---------- Kernel 2: flash attention (async double-buffered K/V tiles in LDS) ----------
// Qh/Kh: [b,h,s,32] f16 (Q pre-scaled). Vt: [b,h,32,s] f16. Obuf: [4096,256] f16.
__global__ __launch_bounds__(256) void attn_flash(
    const _Float16* __restrict__ Qh, const _Float16* __restrict__ Kh,
    const _Float16* __restrict__ Vt, _Float16* __restrict__ Obuf) {
    __shared__ _Float16 kbuf[2][32 * LDT];   // K tile:  rows = kpos, cols = d
    __shared__ _Float16 vbuf[2][32 * LDT];   // V tile:  rows = d,    cols = kpos
    __shared__ _Float16 pbuf[8][16 * LDT];   // per-wave P staging (C-layout -> A-layout)
    int lane = threadIdx.x & 31;
    int wave = threadIdx.x >> 5;
    int lr = lane & 15, hw = lane >> 4;
    int bh = blockIdx.y;                         // b*8 + h
    int q0 = (blockIdx.x * 8 + wave) * 16;       // 16-row Q tile per wave
    const _Float16* Qbase = Qh + ((long)bh * SEQ + q0) * D_HID;
    const _Float16* Kbase = Kh + (long)bh * SEQ * D_HID;
    const _Float16* Vbase = Vt + (long)bh * D_HID * SEQ;
    _Float16* pl = pbuf[wave];

    // one async-copy instruction per wave per tile: waves 0-3 fill K, waves 4-7 fill V
    auto stage = [&](int par, int kt) {
        int t = threadIdx.x;
        if (t < 128) {
            int row = t >> 2, seg = t & 3;   // 32 rows x 4 x 16B
            cp16_to_lds(Kbase + (long)(kt + row) * D_HID + seg * 8,
                        &kbuf[par][row * LDT + seg * 8]);
        } else {
            int t2 = t - 128;
            int row = t2 >> 2, seg = t2 & 3;
            cp16_to_lds(Vbase + (long)row * SEQ + kt + seg * 8,
                        &vbuf[par][row * LDT + seg * 8]);
        }
    };

    v16h aq = load_frag_h16(Qbase, D_HID, lane);   // reused across entire K loop
    v8f o0 = {}, o1 = {};
    float mrow[8], lsum[8];
#pragma unroll
    for (int r = 0; r < 8; ++r) { mrow[r] = -1e30f; lsum[r] = 0.f; }

    stage(0, 0);                                   // prologue fill of buffer 0
    for (int kt = 0; kt < SEQ; kt += 32) {
        int par = (kt >> 5) & 1;
        if (kt + 32 < SEQ) {
            stage(par ^ 1, kt + 32);               // overlap next tile's fill
            WAIT_ASYNC(1);                         // current tile landed (ASYNCcnt <= 1)
        } else {
            WAIT_ASYNC(0);
        }
        __syncthreads();                           // all waves' async fills visible
        const _Float16* kb = kbuf[par];
        const _Float16* vb = vbuf[par];

        // ---- S = Q * K^T for 32 key positions (two 16-col tiles) ----
        v8f z = {};
        v8f s0 = wmma_f16(aq, load_frag_h16(kb, LDT, lane), z);
        v8f s1 = wmma_f16(aq, load_frag_h16(kb + 16 * LDT, LDT, lane), z);

        // ---- online softmax + stage P into LDS in A-fragment source layout ----
#pragma unroll
        for (int r = 0; r < 8; ++r) {
            float tmax = rowmax16(fmaxf(s0[r], s1[r]));
            float mnew = fmaxf(mrow[r], tmax);
            float alpha = __expf(mrow[r] - mnew);
            mrow[r] = mnew;
            o0[r] *= alpha; o1[r] *= alpha;
            float p0 = __expf(s0[r] - mnew);
            float p1 = __expf(s1[r] - mnew);
            lsum[r] = lsum[r] * alpha + rowsum16(p0 + p1);
            int m = r + 8 * hw;                        // row within tile
            pl[m * LDT + lr]      = (_Float16)p0;      // cols kt..kt+15
            pl[m * LDT + 16 + lr] = (_Float16)p1;      // cols kt+16..kt+31
        }
        asm volatile("s_wait_dscnt 0" ::: "memory");   // wave-private LDS: stores -> loads
        v16h ap = load_frag_h16(pl, LDT, lane);        // P as 16x32 A fragment

        // ---- O += P * V (V tile rows are d, cols are kpos) ----
        o0 = wmma_f16(ap, load_frag_h16(vb, LDT, lane), o0);
        o1 = wmma_f16(ap, load_frag_h16(vb + 16 * LDT, LDT, lane), o1);
        __syncthreads();                           // buffer par free for reuse next iter
    }

    // ---- normalize and store [4096,256] f16 (row = b*2048+q, col = h*32+d) ----
    int b = bh >> 3, h = bh & 7;
    _Float16* dst = Obuf + ((long)b * SEQ + q0) * D_MODEL + h * D_HID;
#pragma unroll
    for (int r = 0; r < 8; ++r) {
        float inv = 1.0f / lsum[r];
        int m = r + 8 * hw;
        dst[(long)m * D_MODEL + lr]      = (_Float16)(o0[r] * inv);
        dst[(long)m * D_MODEL + 16 + lr] = (_Float16)(o1[r] * inv);
    }
}

// ---------- Kernel 3: output projection  Y = O @ Wo^T + bo  (fp32 out) ----------
__global__ __launch_bounds__(256) void attn_outproj(
    const _Float16* __restrict__ O, const _Float16* __restrict__ Wo16,
    const float* __restrict__ bo, float* __restrict__ Y) {
    int lane = threadIdx.x & 31;
    int wave = threadIdx.x >> 5;
    int row0 = blockIdx.x * 16;
    int n0 = wave * 32;
    v8f c0 = {}, c1 = {};
#pragma unroll
    for (int k0 = 0; k0 < D_MODEL; k0 += 32) {
        v16h af = load_frag_h16(O + (long)row0 * D_MODEL + k0, D_MODEL, lane);
        v16h b0 = load_frag_h16(Wo16 + (long)n0 * D_MODEL + k0, D_MODEL, lane);
        v16h b1 = load_frag_h16(Wo16 + (long)(n0 + 16) * D_MODEL + k0, D_MODEL, lane);
        c0 = wmma_f16(af, b0, c0);
        c1 = wmma_f16(af, b1, c1);
    }
    int lr = lane & 15, hw = lane >> 4;
#pragma unroll
    for (int r = 0; r < 8; ++r) {
        int m = row0 + r + 8 * hw;
        int col0 = n0 + lr, col1 = n0 + 16 + lr;
        Y[(long)m * D_MODEL + col0] = c0[r] + bo[col0];
        Y[(long)m * D_MODEL + col1] = c1[r] + bo[col1];
    }
}

extern "C" void kernel_launch(void* const* d_in, const int* in_sizes, int n_in,
                              void* d_out, int out_size, void* d_ws, size_t ws_size,
                              hipStream_t stream) {
    const float* query = (const float*)d_in[0];
    const float* key   = (const float*)d_in[1];
    const float* value = (const float*)d_in[2];
    const float* Wq    = (const float*)d_in[3];
    const float* Wk    = (const float*)d_in[4];
    const float* Wv    = (const float*)d_in[5];
    const float* Wo    = (const float*)d_in[6];
    const float* bo    = (const float*)d_in[7];

    // workspace carve-up (halves): weights 4*64K, then Qh/Kh/Vt/Obuf 1M each (~8.5 MB)
    _Float16* w16 = (_Float16*)d_ws;
    _Float16* Wq16 = w16;
    _Float16* Wk16 = w16 + 65536;
    _Float16* Wv16 = w16 + 131072;
    _Float16* Wo16 = w16 + 196608;
    _Float16* Qh   = w16 + 262144;
    _Float16* Kh   = Qh + (long)NROWS * D_MODEL;
    _Float16* Vt   = Kh + (long)NROWS * D_MODEL;
    _Float16* Ob   = Vt + (long)NROWS * D_MODEL;

    attn_cvt_weights<<<256, 256, 0, stream>>>(Wq, Wk, Wv, Wo, w16);

    const float scaling = 0.17677669529663687f;   // 1/sqrt(32), folded into Q
    attn_proj<<<NROWS / 16, 256, 0, stream>>>(query, Wq16, Qh, 0, scaling);
    attn_proj<<<NROWS / 16, 256, 0, stream>>>(key,   Wk16, Kh, 0, 1.0f);
    attn_proj<<<NROWS / 16, 256, 0, stream>>>(value, Wv16, Vt, 1, 1.0f);

    attn_flash<<<dim3(SEQ / 128, BATCH * N_HEAD), 256, 0, stream>>>(Qh, Kh, Vt, Ob);

    attn_outproj<<<NROWS / 16, 256, 0, stream>>>(Ob, Wo16, bo, (float*)d_out);
}
